// MPLayer_28681791603324
// MI455X (gfx1250) — matrix operand order repaired and checked by reading, not compiled, not run on hardware
//
#include <hip/hip_runtime.h>
#include <cstdint>

// ---------------------------------------------------------------------------
// MPNN layer for MI455X (gfx1250): bf16 WMMA edge/node MLPs, async global->LDS
// staging (ASYNCcnt), fp32 atomic message reduction.
// N=50000, E=800000, D=128, DE=16, H=128, MSG_IN=273.
// ---------------------------------------------------------------------------

#define Nn   50000
#define Ee   800000
#define Dd   128
#define DEe  16
#define Hh   128
#define KP   288      // MSG_IN (273) padded to multiple of 32
#define ET   128      // edges per block (8 waves x 16)
#define NT   128      // nodes per block (8 waves x 16)

#define WMT1_E (Hh * KP)          // 36864 u16: Wm1^T [128][288]
#define WMT_E  (WMT1_E + Dd * Hh) // 53248 u16: + Wm2^T [128][128]
#define WUT_E  (2 * Dd * Hh)      // 32768 u16: Wu1^T + Wu2^T

typedef unsigned short u16;
typedef unsigned int   u32;
typedef __attribute__((ext_vector_type(16))) __bf16 v16bf;
typedef __attribute__((ext_vector_type(8)))  float  v8f;

// fp32 -> bf16 (round to nearest even)
__device__ __forceinline__ u16 bf16u(float x) {
  u32 u = __float_as_uint(x);
  u32 r = u + 0x7FFFu + ((u >> 16) & 1u);
  return (u16)(r >> 16);
}
__device__ __forceinline__ u32 pk2(float a, float b) {
  return (u32)bf16u(a) | ((u32)bf16u(b) << 16);
}

// Async 16-byte global -> LDS copy (gfx1250, tracked by ASYNCcnt).
// Generic LDS pointer low 32 bits are the hardware LDS byte address
// (see ISA 10.2).
__device__ __forceinline__ void async_ld16(u16* lds_dst, const void* gsrc) {
  unsigned l = (unsigned)(uintptr_t)lds_dst;
  asm volatile("global_load_async_to_lds_b128 %0, %1, off"
               :: "v"(l), "v"(gsrc) : "memory");
}
__device__ __forceinline__ void wait_async0() {
  asm volatile("s_wait_asynccnt 0" ::: "memory");
}

// A fragment: 16x32 bf16 tile, rows of `tile` (row stride `stride` u16),
// K window [kb, kb+32).  Contiguous k order: half h holds k = kb + 16h + s (same as frag_b).
__device__ __forceinline__ v16bf frag_a(const u16* tile, int stride, int kb, int lane) {
  const int half = lane >> 4;
  const int m    = lane & 15;
  const u16* p = tile + m * stride + kb + half * 16;
  union { uint4 q[2]; v16bf v; } u;
  u.q[0] = *(const uint4*)(p);
  u.q[1] = *(const uint4*)(p + 8);
  return u.v;
}

// B fragment: 32x16 bf16 tile from an N-major (transposed) weight image.
__device__ __forceinline__ v16bf frag_b(const u16* wT, int stride, int kb, int nb, int lane) {
  const int n  = nb + (lane & 15);
  const int ko = kb + ((lane >> 4) << 4);
  const u16* p = wT + n * stride + ko;
  union { uint4 q[2]; v16bf v; } u;
  u.q[0] = *(const uint4*)(p);
  u.q[1] = *(const uint4*)(p + 8);
  return u.v;
}

__device__ __forceinline__ v8f wmma_bf16(v16bf a, v16bf b, v8f c) {
  return __builtin_amdgcn_wmma_f32_16x16x32_bf16(false, a, false, b, (short)0, c,
                                                 false, false);
}

// ---------------------------------------------------------------------------
__global__ void zero_ws(float4* __restrict__ p, int n4) {
  int i = blockIdx.x * blockDim.x + threadIdx.x;
  if (i < n4) p[i] = make_float4(0.f, 0.f, 0.f, 0.f);
}

// One-shot fp32 -> bf16 conversion of node features (coalesced).
__global__ void prep_f(const float4* __restrict__ f4, uint2* __restrict__ fbf2, int n4) {
  int i = blockIdx.x * blockDim.x + threadIdx.x;
  if (i < n4) {
    float4 v = f4[i];
    uint2 pk;
    pk.x = pk2(v.x, v.y);
    pk.y = pk2(v.z, v.w);
    fbf2[i] = pk;
  }
}

// One-shot transpose+convert of all four weight matrices into bf16 N-major
// images (Wm1 zero-padded K 273->288).  Done once per call instead of per
// block: kills the strided per-block fp32 column reads.
__global__ void prep_weights(const float* __restrict__ Wm1, const float* __restrict__ Wm2,
                             const float* __restrict__ Wu1, const float* __restrict__ Wu2,
                             u16* __restrict__ wmT, u16* __restrict__ wuT) {
  int t = blockIdx.x * blockDim.x + threadIdx.x;
  if (t < WMT1_E) {
    int n = t / KP, k = t - n * KP;
    wmT[t] = bf16u(k < 273 ? Wm1[k * Hh + n] : 0.f);
  } else if (t < WMT_E) {
    int i = t - WMT1_E;
    int n = i >> 7, k = i & 127;
    wmT[t] = bf16u(Wm2[k * Dd + n]);
  } else if (t < WMT_E + WUT_E) {
    int i = t - WMT_E;
    if (i < Dd * Hh) {
      int n = i >> 7, k = i & 127;
      wuT[i] = bf16u(Wu1[k * Hh + n]);
    } else {
      int j = i - Dd * Hh;
      int n = j >> 7, k = j & 127;
      wuT[i] = bf16u(Wu2[k * Dd + n]);
    }
  }
}

// ---------------------------------------------------------------------------
// Edge kernel: async-stage bf16 weights + gathered bf16 msg_in tile in LDS,
// then per-wave 16x288 @ 288x128 (relu) @ 128x128, scatter-add to m_sum[dst].
// LDS: W1T 72KB + W2T 32KB + msg 72KB + hid 32KB = 208KB (dynamic).
__global__ void __launch_bounds__(256)
edge_kernel(const u16* __restrict__ fbf, const float* __restrict__ x,
            const float* __restrict__ w, const int* __restrict__ src,
            const int* __restrict__ dst, const u16* __restrict__ wmT,
            const float* __restrict__ bm1, const float* __restrict__ bm2,
            float* __restrict__ msum) {
  extern __shared__ u16 smem[];
  u16* sW1  = smem;                    // [128][KP]  Wm1^T  (bf16)
  u16* sW2  = sW1 + WMT1_E;            // [128][128] Wm2^T
  u16* sMsg = sW2 + Dd * Hh;           // [ET][KP]   gathered msg_in
  u16* sHid = sMsg + ET * KP;          // [ET][128]  relu hidden

  const int tid = threadIdx.x;
  const int e0  = blockIdx.x * ET;

  // --- async bulk copy of pre-converted weights (contiguous, 106496 B)
  for (int t = tid; t < WMT_E / 8; t += 256)          // 6656 x 16B, 26/thread
    async_ld16(smem + t * 8, wmT + t * 8);

  // --- async gather of bf16 feature rows: cols 0..127 <- fbf[src], 128..255 <- fbf[dst]
  for (int t = tid; t < ET * 32; t += 256) {          // 4096 x 16B, 16/thread
    int el = t >> 5, c = t & 31;
    int e = e0 + el;
    int row = (c < 16) ? src[e] : dst[e];
    async_ld16(sMsg + el * KP + c * 8, fbf + (size_t)row * Dd + (c & 15) * 8);
  }

  // --- tail: edge features (convert), squared distance, zero pad
  if (tid < ET) {
    int e = e0 + tid;
    u16* p = sMsg + tid * KP + 256;
    const float4* wr = (const float4*)(w + (size_t)e * DEe);
    #pragma unroll
    for (int c = 0; c < 4; ++c) {
      float4 v = wr[c];
      uint2 pk;
      pk.x = pk2(v.x, v.y);
      pk.y = pk2(v.z, v.w);
      *(uint2*)(p + 4 * c) = pk;                      // cols 256..271
    }
    int s = src[e], d = dst[e];
    float dx = x[3 * s + 0] - x[3 * d + 0];
    float dy = x[3 * s + 1] - x[3 * d + 1];
    float dz = x[3 * s + 2] - x[3 * d + 2];
    uint4 q;
    q.x = pk2(dx * dx + dy * dy + dz * dz, 0.f);
    q.y = 0u; q.z = 0u; q.w = 0u;
    *(uint4*)(p + 16) = q;                            // cols 272..279
    *(uint4*)(p + 24) = make_uint4(0u, 0u, 0u, 0u);   // cols 280..287
  }
  wait_async0();
  __syncthreads();

  const int lane = tid & 31;
  const int wv   = tid >> 5;
  const int r0   = wv * 16;
  const int half = lane >> 4;
  const int nlo  = lane & 15;

  // --- GEMM1: [16 x 288] @ [288 x 128]
  v8f acc[8];
  #pragma unroll
  for (int i = 0; i < 8; ++i) acc[i] = (v8f){0.f,0.f,0.f,0.f,0.f,0.f,0.f,0.f};
  #pragma unroll
  for (int kb = 0; kb < KP; kb += 32) {
    v16bf a = frag_a(sMsg + r0 * KP, KP, kb, lane);
    #pragma unroll
    for (int nt = 0; nt < 8; ++nt) {
      v16bf b = frag_b(sW1, KP, kb, nt * 16, lane);
      acc[nt] = wmma_bf16(a, b, acc[nt]);
    }
  }

  // --- bias + relu -> bf16 hidden tile (same-wave LDS, DS in-order)
  #pragma unroll
  for (int nt = 0; nt < 8; ++nt) {
    float bb = bm1[nt * 16 + nlo];
    #pragma unroll
    for (int r = 0; r < 8; ++r) {
      int m = r + half * 8;
      float v = acc[nt][r] + bb;
      v = v > 0.f ? v : 0.f;
      sHid[(r0 + m) * Hh + nt * 16 + nlo] = bf16u(v);
    }
  }

  // --- GEMM2: [16 x 128] @ [128 x 128]
  v8f acc2[8];
  #pragma unroll
  for (int i = 0; i < 8; ++i) acc2[i] = (v8f){0.f,0.f,0.f,0.f,0.f,0.f,0.f,0.f};
  #pragma unroll
  for (int kb = 0; kb < Hh; kb += 32) {
    v16bf a = frag_a(sHid + r0 * Hh, Hh, kb, lane);
    #pragma unroll
    for (int nt = 0; nt < 8; ++nt) {
      v16bf b = frag_b(sW2, Hh, kb, nt * 16, lane);
      acc2[nt] = wmma_bf16(a, b, acc2[nt]);
    }
  }

  // --- bias + fp32 atomic scatter to m_sum[dst[e]]
  #pragma unroll
  for (int nt = 0; nt < 8; ++nt) {
    float bb = bm2[nt * 16 + nlo];
    #pragma unroll
    for (int r = 0; r < 8; ++r) {
      int m = r + half * 8;
      int e = e0 + r0 + m;
      int row = dst[e];
      float v = acc2[nt][r] + bb;
      __hip_atomic_fetch_add(&msum[(size_t)row * Dd + nt * 16 + nlo], v,
                             __ATOMIC_RELAXED, __HIP_MEMORY_SCOPE_AGENT);
    }
  }
}

// ---------------------------------------------------------------------------
// Node kernel: h = m_sum + f ; out = relu(h@Wu1+bu1)@Wu2 + bu2.
// LDS: U1T 32KB + U2T 32KB + h 32KB + hid 32KB = 128KB (dynamic).
__global__ void __launch_bounds__(256)
node_kernel(const float* __restrict__ f, const float* __restrict__ msum,
            const u16* __restrict__ wuT, const float* __restrict__ bu1,
            const float* __restrict__ bu2, float* __restrict__ out, int nN) {
  extern __shared__ u16 smem[];
  u16* sU1  = smem;                    // [128][128] Wu1^T
  u16* sU2  = sU1 + Dd * Hh;           // [128][128] Wu2^T
  u16* sH   = sU2 + Hh * Dd;           // [NT][128]  h tile
  u16* sHid = sH  + NT * Dd;           // [NT][128]  relu hidden

  const int tid = threadIdx.x;
  const int n0  = blockIdx.x * NT;

  // --- async bulk copy of pre-converted Wu1^T|Wu2^T (65536 B)
  for (int t = tid; t < WUT_E / 8; t += 256)          // 4096 x 16B, 16/thread
    async_ld16(smem + t * 8, wuT + t * 8);

  // --- h tile: m_sum + f in fp32, convert to bf16
  for (int t = tid; t < NT * 32; t += 256) {
    int rl = t >> 5, c = t & 31;
    int node = n0 + rl;
    if (node >= nN) node = 0;                         // safe clamp (partial tile)
    float4 a = ((const float4*)(f    + (size_t)node * Dd))[c];
    float4 b = ((const float4*)(msum + (size_t)node * Dd))[c];
    uint2 pk;
    pk.x = pk2(a.x + b.x, a.y + b.y);
    pk.y = pk2(a.z + b.z, a.w + b.w);
    *(uint2*)(sH + rl * Dd + 4 * c) = pk;
  }
  wait_async0();
  __syncthreads();

  const int lane = tid & 31;
  const int wv   = tid >> 5;
  const int r0   = wv * 16;
  const int half = lane >> 4;
  const int nlo  = lane & 15;

  v8f acc[8];
  #pragma unroll
  for (int i = 0; i < 8; ++i) acc[i] = (v8f){0.f,0.f,0.f,0.f,0.f,0.f,0.f,0.f};
  #pragma unroll
  for (int kb = 0; kb < Dd; kb += 32) {
    v16bf a = frag_a(sH + r0 * Dd, Dd, kb, lane);
    #pragma unroll
    for (int nt = 0; nt < 8; ++nt) {
      v16bf b = frag_b(sU1, Dd, kb, nt * 16, lane);
      acc[nt] = wmma_bf16(a, b, acc[nt]);
    }
  }
  #pragma unroll
  for (int nt = 0; nt < 8; ++nt) {
    float bb = bu1[nt * 16 + nlo];
    #pragma unroll
    for (int r = 0; r < 8; ++r) {
      int m = r + half * 8;
      float v = acc[nt][r] + bb;
      v = v > 0.f ? v : 0.f;
      sHid[(r0 + m) * Hh + nt * 16 + nlo] = bf16u(v);
    }
  }

  v8f acc2[8];
  #pragma unroll
  for (int i = 0; i < 8; ++i) acc2[i] = (v8f){0.f,0.f,0.f,0.f,0.f,0.f,0.f,0.f};
  #pragma unroll
  for (int kb = 0; kb < Hh; kb += 32) {
    v16bf a = frag_a(sHid + r0 * Hh, Hh, kb, lane);
    #pragma unroll
    for (int nt = 0; nt < 8; ++nt) {
      v16bf b = frag_b(sU2, Hh, kb, nt * 16, lane);
      acc2[nt] = wmma_bf16(a, b, acc2[nt]);
    }
  }
  #pragma unroll
  for (int nt = 0; nt < 8; ++nt) {
    float bb = bu2[nt * 16 + nlo];
    #pragma unroll
    for (int r = 0; r < 8; ++r) {
      int m = r + half * 8;
      int node = n0 + r0 + m;
      if (node < nN)
        out[(size_t)node * Dd + nt * 16 + nlo] = acc2[nt][r] + bb;
    }
  }
}

// ---------------------------------------------------------------------------
extern "C" void kernel_launch(void* const* d_in, const int* in_sizes, int n_in,
                              void* d_out, int out_size, void* d_ws, size_t ws_size,
                              hipStream_t stream) {
  const float* f   = (const float*)d_in[0];
  const float* x   = (const float*)d_in[1];
  const float* w   = (const float*)d_in[2];
  const int*   src = (const int*)d_in[3];
  const int*   dst = (const int*)d_in[4];
  const float* Wm1 = (const float*)d_in[5];
  const float* bm1 = (const float*)d_in[6];
  const float* Wm2 = (const float*)d_in[7];
  const float* bm2 = (const float*)d_in[8];
  const float* Wu1 = (const float*)d_in[9];
  const float* bu1 = (const float*)d_in[10];
  const float* Wu2 = (const float*)d_in[11];
  const float* bu2 = (const float*)d_in[12];
  float* out = (float*)d_out;

  // Workspace layout (bytes, all 16B aligned):
  //   [0)            msum  : N*D fp32   = 25,600,000
  //   [25,600,000)   fbf   : N*D bf16   = 12,800,000
  //   [38,400,000)   wmT   : Wm1^T|Wm2^T bf16 = 106,496
  //   [38,506,496)   wuT   : Wu1^T|Wu2^T bf16 =  65,536
  char* wsb  = (char*)d_ws;
  float* msum = (float*)wsb;
  u16*   fbf  = (u16*)(wsb + (size_t)Nn * Dd * 4);
  u16*   wmT  = (u16*)(wsb + (size_t)Nn * Dd * 4 + (size_t)Nn * Dd * 2);
  u16*   wuT  = (u16*)((char*)wmT + (size_t)WMT_E * 2);

  const int edge_lds = (WMT_E + ET * KP + ET * Hh) * (int)sizeof(u16); // 212992
  const int node_lds = (WUT_E + NT * Dd * 2) * (int)sizeof(u16);      // 131072
  hipFuncSetAttribute(reinterpret_cast<const void*>(&edge_kernel),
                      hipFuncAttributeMaxDynamicSharedMemorySize, edge_lds);
  hipFuncSetAttribute(reinterpret_cast<const void*>(&node_kernel),
                      hipFuncAttributeMaxDynamicSharedMemorySize, node_lds);

  const int n4 = Nn * Dd / 4;
  prep_f<<<(n4 + 255) / 256, 256, 0, stream>>>((const float4*)f, (uint2*)fbf, n4);
  prep_weights<<<(WMT_E + WUT_E + 255) / 256, 256, 0, stream>>>(Wm1, Wm2, Wu1, Wu2,
                                                                wmT, wuT);
  zero_ws<<<(n4 + 255) / 256, 256, 0, stream>>>((float4*)msum, n4);
  edge_kernel<<<Ee / ET, 256, edge_lds, stream>>>(fbf, x, w, src, dst, wmT,
                                                  bm1, bm2, msum);
  node_kernel<<<(Nn + NT - 1) / NT, 256, node_lds, stream>>>(f, msum, wuT,
                                                             bu1, bu2, out, Nn);
}
